// MaskedAttentionWithRoPE_65927747993850
// MI455X (gfx1250) — matrix operand order repaired and checked by reading, not compiled, not run on hardware
//
#include <hip/hip_runtime.h>
#include <hip/hip_bf16.h>

typedef __attribute__((ext_vector_type(16))) _Float16 v16h;
typedef __attribute__((ext_vector_type(8)))  _Float16 v8h;
typedef __attribute__((ext_vector_type(8)))  float    v8f;

#define DIMN   1024
#define NHEAD  16
#define HDIM   64
#define BATCH  2
#define SEQ    2048
#define NTOK   (BATCH*SEQ)   // 4096
#define LDST   48            // padded LDS row stride (halves); 96B keeps 16B alignment

#define WMMA_F32_F16(a, b, c) \
  __builtin_amdgcn_wmma_f32_16x16x32_f16(false, (a), false, (b), (short)0, (c), false, false)

// ---- fragment helpers (CDNA5 wave32 WMMA layouts, CDNA5 ISA) ----

__device__ __forceinline__ v16h make_frag(v8h lo, v8h hi) {
  return __builtin_shufflevector(lo, hi, 0,1,2,3,4,5,6,7,8,9,10,11,12,13,14,15);
}

// A fragment 16x32 (16-bit): lane = M row; elements 0..15 = K[16*hg .. +16) (this code's contiguous k order; B uses the same).
// rowp points at A[row][0], k0 = K offset.
__device__ __forceinline__ v16h load_a_frag(const _Float16* rowp, int k0, int hg) {
  v8h c0 = *(const v8h*)(rowp + k0 + 16*hg);
  v8h c1 = *(const v8h*)(rowp + k0 + 16*hg + 8);
  return make_frag(c0, c1);
}

// ---------------------------------------------------------------------------
// Prep: f32 -> f16 convert, and weight transpose (W[k][n] -> Wt[n][k]) to f16
// ---------------------------------------------------------------------------
__global__ void cvt_f16_kernel(const float* __restrict__ s, _Float16* __restrict__ d, int n) {
  int i = blockIdx.x * blockDim.x + threadIdx.x;
  if (i < n) d[i] = (_Float16)s[i];
}

__global__ void transpose_f16_kernel(const float* __restrict__ w, _Float16* __restrict__ wt) {
  int i = blockIdx.x * blockDim.x + threadIdx.x;   // over DIMN*DIMN
  int k = i >> 10, n = i & (DIMN - 1);
  wt[(size_t)n * DIMN + k] = (_Float16)w[i];
}

// ---------------------------------------------------------------------------
// QKV projection + fused RoPE. grid=(N/64, M/128, 3), block=128 (4 waves).
// Each wave: one 32(M) x 64(N) tile, K-loop step 32, 2 A-frags x 4 B-frags
// -> 8 WMMA per K-step (each B fragment reused by two WMMAs).
// z=0 -> Q [b,h,s,d] (+RoPE), z=1 -> K [b,h,s,d] (+RoPE), z=2 -> V^T [b,h,d,s]
// ---------------------------------------------------------------------------
__global__ void __launch_bounds__(128)
qkv_rope_kernel(const _Float16* __restrict__ xh,
                const _Float16* __restrict__ wqt, const _Float16* __restrict__ wkt,
                const _Float16* __restrict__ wvt,
                const float* __restrict__ bq, const float* __restrict__ bk,
                const float* __restrict__ bv,
                const float* __restrict__ fc, const float* __restrict__ fs,
                _Float16* __restrict__ Qh, _Float16* __restrict__ Kh,
                _Float16* __restrict__ Vt) {
  const int lane = threadIdx.x & 31;
  const int wave = threadIdx.x >> 5;
  const int hg   = lane >> 4;       // lane half-group
  const int lr   = lane & 15;
  const int z    = blockIdx.z;

  const _Float16* wt  = (z == 0) ? wqt : (z == 1) ? wkt : wvt;
  const float*    bia = (z == 0) ? bq  : (z == 1) ? bk  : bv;

  const int Mbase = (blockIdx.y * 4 + wave) * 32;
  const int Nbase = blockIdx.x * 64;

  v8f acc[2][4] = {};
  const _Float16* arow0 = xh + (size_t)(Mbase + lr) * DIMN;
  const _Float16* arow1 = arow0 + (size_t)16 * DIMN;

  for (int k0 = 0; k0 < DIMN; k0 += 32) {
    __builtin_prefetch(arow0 + k0 + 128, 0, 3);
    __builtin_prefetch(arow1 + k0 + 128, 0, 3);
    v16h a0 = load_a_frag(arow0, k0, hg);
    v16h a1 = load_a_frag(arow1, k0, hg);
    v16h b0 = *(const v16h*)(wt + (size_t)(Nbase +  0 + lr) * DIMN + k0 + 16*hg);
    v16h b1 = *(const v16h*)(wt + (size_t)(Nbase + 16 + lr) * DIMN + k0 + 16*hg);
    v16h b2 = *(const v16h*)(wt + (size_t)(Nbase + 32 + lr) * DIMN + k0 + 16*hg);
    v16h b3 = *(const v16h*)(wt + (size_t)(Nbase + 48 + lr) * DIMN + k0 + 16*hg);
    acc[0][0] = WMMA_F32_F16(a0, b0, acc[0][0]);
    acc[1][0] = WMMA_F32_F16(a1, b0, acc[1][0]);
    acc[0][1] = WMMA_F32_F16(a0, b1, acc[0][1]);
    acc[1][1] = WMMA_F32_F16(a1, b1, acc[1][1]);
    acc[0][2] = WMMA_F32_F16(a0, b2, acc[0][2]);
    acc[1][2] = WMMA_F32_F16(a1, b2, acc[1][2]);
    acc[0][3] = WMMA_F32_F16(a0, b3, acc[0][3]);
    acc[1][3] = WMMA_F32_F16(a1, b3, acc[1][3]);
  }

  // Epilogue: C layout => row M = rg*16 + r + 8*hg, col N = lr (per slice)
#pragma unroll
  for (int rg = 0; rg < 2; rg++) {
#pragma unroll
    for (int j = 0; j < 4; j++) {
      const int n  = Nbase + 16*j + lr;
      const float bn = bia[n];
      const int h = n >> 6, d = n & 63;
#pragma unroll
      for (int r = 0; r < 8; r++) {
        const int t  = Mbase + rg*16 + r + 8*hg;
        const int bb = t >> 11;            // / SEQ
        const int s  = t & (SEQ - 1);
        float v = acc[rg][j][r] + bn;
        if (z < 2) {
          // RoPE: even/odd feature pair lives in adjacent lanes (same row, same i)
          const int i = d >> 1;
          float pv = __shfl_xor(v, 1, 32);
          float c  = fc[s * 32 + i];
          float sn = fs[s * 32 + i];
          v = ((d & 1) == 0) ? (v * c - pv * sn) : (pv * sn + v * c);
          _Float16* dst = (z == 0) ? Qh : Kh;
          dst[(((size_t)(bb * NHEAD + h)) * SEQ + s) * HDIM + d] = (_Float16)v;
        } else {
          Vt[(((size_t)(bb * NHEAD + h)) * HDIM + d) * SEQ + s] = (_Float16)v;
        }
      }
    }
  }
}

// ---------------------------------------------------------------------------
// Flash attention. grid=(SEQ/64, NHEAD, BATCH), block=128 (4 waves).
// Each wave: one 16-query tile, online softmax over 32 keys per iteration.
// ---------------------------------------------------------------------------
__global__ void __launch_bounds__(128)
flash_attn_kernel(const _Float16* __restrict__ Qh, const _Float16* __restrict__ Kh,
                  const _Float16* __restrict__ Vt, const float* __restrict__ mask,
                  _Float16* __restrict__ AO) {
  __shared__ _Float16 lds[4 * 16 * LDST];
  const int lane = threadIdx.x & 31;
  const int wave = threadIdx.x >> 5;
  const int hg = lane >> 4, lr = lane & 15;
  const int qt = blockIdx.x * 4 + wave;
  const int h  = blockIdx.y, bb = blockIdx.z;
  const size_t bh = (size_t)bb * NHEAD + h;

  const _Float16* Qb = Qh + bh * SEQ * HDIM;
  const _Float16* Kb = Kh + bh * SEQ * HDIM;
  const _Float16* Vb = Vt + bh * HDIM * SEQ;
  _Float16* myLds = lds + wave * 16 * LDST;

  // resident Q A-fragments (d = 0..31 and 32..63)
  const _Float16* qrow = Qb + (size_t)(qt * 16 + lr) * HDIM;
  const v16h qa0 = load_a_frag(qrow, 0, hg);
  const v16h qa1 = load_a_frag(qrow, 32, hg);

  v8f o[4] = {};
  float mrow[8], lsum[8];
#pragma unroll
  for (int r = 0; r < 8; r++) { mrow[r] = -1e30f; lsum[r] = 0.0f; }

  const float scale = 0.125f;   // 1/sqrt(64)

  for (int kt = 0; kt < SEQ; kt += 32) {
    // scores: S = Q(16x64) * K^T(64x32); B-frag lane = key col, elems = d
    v8f s0 = {}, s1 = {};
    {
      const _Float16* k0p = Kb + (size_t)(kt + lr) * HDIM + 16*hg;
      const _Float16* k1p = Kb + (size_t)(kt + 16 + lr) * HDIM + 16*hg;
      __builtin_prefetch(k0p + 32 * HDIM, 0, 3);
      v16h b00 = *(const v16h*)(k0p);
      v16h b01 = *(const v16h*)(k0p + 32);
      v16h b10 = *(const v16h*)(k1p);
      v16h b11 = *(const v16h*)(k1p + 32);
      s0 = WMMA_F32_F16(qa0, b00, s0);
      s0 = WMMA_F32_F16(qa1, b01, s0);
      s1 = WMMA_F32_F16(qa0, b10, s1);
      s1 = WMMA_F32_F16(qa1, b11, s1);
    }

    // online softmax (rows: reg r + lane half; cols: 16 lanes of the half)
#pragma unroll
    for (int r = 0; r < 8; r++) {
      const int q = qt * 16 + r + 8*hg;
      float a0 = s0[r] * scale + mask[(size_t)q * SEQ + kt + lr];
      float a1 = s1[r] * scale + mask[(size_t)q * SEQ + kt + 16 + lr];
      float mx = fmaxf(a0, a1);
      mx = fmaxf(mx, __shfl_xor(mx, 1, 32));
      mx = fmaxf(mx, __shfl_xor(mx, 2, 32));
      mx = fmaxf(mx, __shfl_xor(mx, 4, 32));
      mx = fmaxf(mx, __shfl_xor(mx, 8, 32));
      const float mnew  = fmaxf(mrow[r], mx);
      const float alpha = __expf(mrow[r] - mnew);
      mrow[r] = mnew;
      const float e0 = __expf(a0 - mnew);
      const float e1 = __expf(a1 - mnew);
      float rs = e0 + e1;
      rs += __shfl_xor(rs, 1, 32);
      rs += __shfl_xor(rs, 2, 32);
      rs += __shfl_xor(rs, 4, 32);
      rs += __shfl_xor(rs, 8, 32);
      lsum[r] = lsum[r] * alpha + rs;
#pragma unroll
      for (int j = 0; j < 4; j++) o[j][r] *= alpha;
      // P (f32, C layout) -> LDS as f16 at [row][col]
      const int row = r + 8*hg;
      myLds[row * LDST + lr]      = (_Float16)e0;
      myLds[row * LDST + 16 + lr] = (_Float16)e1;
    }

    // same-wave LDS store->load ordering (CDNA5 split counters)
    asm volatile("s_wait_dscnt 0x0" ::: "memory");

    // re-read P as A fragment 16x32
    const v16h pa = make_frag(*(const v8h*)(myLds + lr * LDST + 16*hg),
                              *(const v8h*)(myLds + lr * LDST + 16*hg + 8));
    // O += P * V ; V^T layout [d][s]: B-frag lane = d col, elems = contiguous keys
#pragma unroll
    for (int j = 0; j < 4; j++) {
      v16h vb = *(const v16h*)(Vb + (size_t)(16*j + lr) * SEQ + kt + 16*hg);
      o[j] = WMMA_F32_F16(pa, vb, o[j]);
    }
  }

  // normalize and store to [b, q, h*64+d] as f16 (A operand of out-proj)
#pragma unroll
  for (int j = 0; j < 4; j++) {
#pragma unroll
    for (int r = 0; r < 8; r++) {
      const int q = qt * 16 + r + 8*hg;
      const float val = o[j][r] / lsum[r];
      const size_t t = (size_t)bb * SEQ + q;
      AO[t * DIMN + h * HDIM + 16*j + lr] = (_Float16)val;
    }
  }
}

// ---------------------------------------------------------------------------
// Output projection: out = AO(4096x1024 f16) * Wo (via Wo^T f16), f32 result.
// Same 32x64 wave tile as the QKV GEMM.
// ---------------------------------------------------------------------------
__global__ void __launch_bounds__(128)
oproj_kernel(const _Float16* __restrict__ A, const _Float16* __restrict__ wot,
             float* __restrict__ out) {
  const int lane = threadIdx.x & 31;
  const int wave = threadIdx.x >> 5;
  const int hg = lane >> 4, lr = lane & 15;
  const int Mbase = (blockIdx.y * 4 + wave) * 32;
  const int Nbase = blockIdx.x * 64;

  v8f acc[2][4] = {};
  const _Float16* arow0 = A + (size_t)(Mbase + lr) * DIMN;
  const _Float16* arow1 = arow0 + (size_t)16 * DIMN;

  for (int k0 = 0; k0 < DIMN; k0 += 32) {
    __builtin_prefetch(arow0 + k0 + 128, 0, 3);
    __builtin_prefetch(arow1 + k0 + 128, 0, 3);
    v16h a0 = load_a_frag(arow0, k0, hg);
    v16h a1 = load_a_frag(arow1, k0, hg);
    v16h b0 = *(const v16h*)(wot + (size_t)(Nbase +  0 + lr) * DIMN + k0 + 16*hg);
    v16h b1 = *(const v16h*)(wot + (size_t)(Nbase + 16 + lr) * DIMN + k0 + 16*hg);
    v16h b2 = *(const v16h*)(wot + (size_t)(Nbase + 32 + lr) * DIMN + k0 + 16*hg);
    v16h b3 = *(const v16h*)(wot + (size_t)(Nbase + 48 + lr) * DIMN + k0 + 16*hg);
    acc[0][0] = WMMA_F32_F16(a0, b0, acc[0][0]);
    acc[1][0] = WMMA_F32_F16(a1, b0, acc[1][0]);
    acc[0][1] = WMMA_F32_F16(a0, b1, acc[0][1]);
    acc[1][1] = WMMA_F32_F16(a1, b1, acc[1][1]);
    acc[0][2] = WMMA_F32_F16(a0, b2, acc[0][2]);
    acc[1][2] = WMMA_F32_F16(a1, b2, acc[1][2]);
    acc[0][3] = WMMA_F32_F16(a0, b3, acc[0][3]);
    acc[1][3] = WMMA_F32_F16(a1, b3, acc[1][3]);
  }

#pragma unroll
  for (int rg = 0; rg < 2; rg++) {
#pragma unroll
    for (int j = 0; j < 4; j++) {
      const int n = Nbase + 16*j + lr;
#pragma unroll
      for (int r = 0; r < 8; r++) {
        const int t = Mbase + rg*16 + r + 8*hg;
        out[(size_t)t * DIMN + n] = acc[rg][j][r];
      }
    }
  }
}

// ---------------------------------------------------------------------------
extern "C" void kernel_launch(void* const* d_in, const int* in_sizes, int n_in,
                              void* d_out, int out_size, void* d_ws, size_t ws_size,
                              hipStream_t stream) {
  const float* x    = (const float*)d_in[0];
  const float* mask = (const float*)d_in[1];
  const float* fc   = (const float*)d_in[2];
  const float* fs   = (const float*)d_in[3];
  const float* wq   = (const float*)d_in[4];
  const float* bq   = (const float*)d_in[5];
  const float* wk   = (const float*)d_in[6];
  const float* bk   = (const float*)d_in[7];
  const float* wv   = (const float*)d_in[8];
  const float* bv   = (const float*)d_in[9];
  const float* wo   = (const float*)d_in[10];
  float* out = (float*)d_out;

  _Float16* p   = (_Float16*)d_ws;
  _Float16* xh  = p; p += (size_t)NTOK * DIMN;
  _Float16* wqt = p; p += (size_t)DIMN * DIMN;
  _Float16* wkt = p; p += (size_t)DIMN * DIMN;
  _Float16* wvt = p; p += (size_t)DIMN * DIMN;
  _Float16* wot = p; p += (size_t)DIMN * DIMN;
  _Float16* Qh  = p; p += (size_t)NTOK * DIMN;   // [b,h,s,d]
  _Float16* Kh  = p; p += (size_t)NTOK * DIMN;   // [b,h,s,d]
  _Float16* Vt  = p; p += (size_t)NTOK * DIMN;   // [b,h,d,s]
  _Float16* AO  = p; p += (size_t)NTOK * DIMN;   // [b,s,h*d]

  const int nxe = NTOK * DIMN;
  cvt_f16_kernel<<<nxe / 256, 256, 0, stream>>>(x, xh, nxe);
  const int nwe = DIMN * DIMN;
  transpose_f16_kernel<<<nwe / 256, 256, 0, stream>>>(wq, wqt);
  transpose_f16_kernel<<<nwe / 256, 256, 0, stream>>>(wk, wkt);
  transpose_f16_kernel<<<nwe / 256, 256, 0, stream>>>(wv, wvt);
  transpose_f16_kernel<<<nwe / 256, 256, 0, stream>>>(wo, wot);

  dim3 g1(DIMN / 64, NTOK / 128, 3);         // (16, 32, 3)
  qkv_rope_kernel<<<g1, 128, 0, stream>>>(xh, wqt, wkt, wvt, bq, bk, bv,
                                          fc, fs, Qh, Kh, Vt);

  dim3 g2(SEQ / 64, NHEAD, BATCH);           // (32, 16, 2)
  flash_attn_kernel<<<g2, 128, 0, stream>>>(Qh, Kh, Vt, mask, AO);

  dim3 g3(DIMN / 64, NTOK / 128, 1);         // (16, 32, 1)
  oproj_kernel<<<g3, 128, 0, stream>>>(AO, wot, out);
}
